// AttentionBlock_11201274708432
// MI455X (gfx1250) — matrix operand order repaired and checked by reading, not compiled, not run on hardware
//
#include <hip/hip_runtime.h>
#include <hip/hip_bf16.h>
#include <stdint.h>

typedef _Float16 h16;
typedef __attribute__((ext_vector_type(16))) _Float16 v16h;
typedef __attribute__((ext_vector_type(8)))  _Float16 v8h;
typedef __attribute__((ext_vector_type(8)))  float    v8f;
typedef __attribute__((ext_vector_type(4)))  unsigned int u32x4;
typedef __attribute__((ext_vector_type(8)))  unsigned int u32x8;

#define C_DIM 1024
#define L_DIM 2048
#define NHEAD 16
#define DH    64
#define OC3   3072   // 3*C
#define BIAS_PITCH 4096

// ---------------------------------------------------------------------------
// WMMA helpers (CDNA5 wave32, 16x16x32 f16 -> f32)
// ---------------------------------------------------------------------------
static __device__ __forceinline__ v8f wmma_f16f32(v16h a, v16h b, v8f c) {
  return __builtin_amdgcn_wmma_f32_16x16x32_f16(false, a, false, b, (short)0, c,
                                                false, false);
}

// A fragment: 16x32 (MxK), row-major memory, K contiguous, row stride rs (elems).
static __device__ __forceinline__ v16h load_A_frag(const h16* base, int rs) {
  int lane = threadIdx.x & 31;
  int m = lane & 15;
  int hi = lane >> 4;
  const h16* r = base + m * rs;
  v8h lo = *(const v8h*)(r + hi * 8);
  v8h hh = *(const v8h*)(r + 16 + hi * 8);
  v16h a;
  #pragma unroll
  for (int i = 0; i < 8; ++i) { a[i] = lo[i]; a[i + 8] = hh[i]; }
  return a;
}

// B fragment: 32x16 (KxN), memory [N][K] K-contiguous, col stride cs (elems).
static __device__ __forceinline__ v16h load_B_frag(const h16* base, int cs) {
  int lane = threadIdx.x & 31;
  int n = lane & 15;
  int hi = lane >> 4;
  const h16* c = base + n * cs + hi * 8;
  v8h lo = *(const v8h*)(c);
  v8h hh = *(const v8h*)(c + 16);
  v16h b;
  #pragma unroll
  for (int i = 0; i < 8; ++i) { b[i] = lo[i]; b[i + 8] = hh[i]; }
  return b;
}

// ---------------------------------------------------------------------------
// Tensor Data Mover: 2D tile global -> LDS (ISA 08_async_tensor D# layout).
// 2-group form (VADDR2/3 = NULL, tensors up to 2D). data_size = 2 bytes.
// All arguments must be wave-uniform. EXEC is ignored by tensor ops.
// ---------------------------------------------------------------------------
static __device__ __forceinline__ void tdm_load_2d(unsigned lds_addr,
                                                   const h16* gaddr,
                                                   unsigned tile_d0,   // elems, contiguous dim
                                                   unsigned tile_d1,   // rows
                                                   unsigned long long stride) { // elems
  unsigned long long ga = (unsigned long long)(uintptr_t)gaddr;
  u32x4 g0;
  g0[0] = 1u;                                              // count=1, user desc
  g0[1] = lds_addr;                                        // [63:32] lds_addr (bytes)
  g0[2] = (unsigned)ga;                                    // global_addr[31:0]
  g0[3] = (unsigned)((ga >> 32) & 0x1FFFFFFu) | (2u << 30); // ga[56:32] | type=2
  u32x8 g1;
  g1[0] = 0x00010000u;                                     // data_size=1 (2B), mask=0
  g1[1] = (tile_d0 & 0xFFFFu) << 16;                       // tensor_dim0[15:0] @ [63:48]
  g1[2] = (tile_d0 >> 16) | ((tile_d1 & 0xFFFFu) << 16);   // td0 hi | tensor_dim1 lo
  g1[3] = (tile_d1 >> 16) | (tile_d0 << 16);               // td1 hi | tile_dim0
  g1[4] = tile_d1 & 0xFFFFu;                               // tile_dim1 (tile_dim2=0)
  g1[5] = (unsigned)(stride & 0xFFFFFFFFu);                // tensor_dim0_stride[31:0]
  g1[6] = (unsigned)((stride >> 32) & 0xFFFFu);            // stride[47:32]
  g1[7] = 0u;
  asm volatile("tensor_load_to_lds %0, %1" :: "s"(g0), "s"(g1) : "memory");
}

// ---------------------------------------------------------------------------
// GroupNorm stats (one block per group: 32 channels x 2048)
// ---------------------------------------------------------------------------
__global__ void gn_stats(const float* __restrict__ x, float* __restrict__ mean,
                         float* __restrict__ rstd) {
  int g = blockIdx.x;
  const float* p = x + (size_t)g * 32 * L_DIM;
  float s = 0.f, sq = 0.f;
  for (int i = threadIdx.x; i < 32 * L_DIM; i += blockDim.x) {
    float v = p[i];
    s += v; sq += v * v;
  }
  __shared__ float sh[256], sh2[256];
  sh[threadIdx.x] = s; sh2[threadIdx.x] = sq;
  __syncthreads();
  for (int st = 128; st > 0; st >>= 1) {
    if ((int)threadIdx.x < st) {
      sh[threadIdx.x] += sh[threadIdx.x + st];
      sh2[threadIdx.x] += sh2[threadIdx.x + st];
    }
    __syncthreads();
  }
  if (threadIdx.x == 0) {
    float m = sh[0] * (1.0f / 65536.0f);
    float var = sh2[0] * (1.0f / 65536.0f) - m * m;
    mean[g] = m;
    rstd[g] = rsqrtf(var + 1e-5f);
  }
}

__global__ void gn_apply(const float* __restrict__ x, const float* __restrict__ gw,
                         const float* __restrict__ gb, const float* __restrict__ mean,
                         const float* __restrict__ rstd, h16* __restrict__ xnt) {
  int idx = blockIdx.x * blockDim.x + threadIdx.x;  // idx = l*1024 + c
  if (idx >= C_DIM * L_DIM) return;
  int l = idx >> 10;
  int c = idx & 1023;
  int g = c >> 5;
  float v = (x[(size_t)c * L_DIM + l] - mean[g]) * rstd[g] * gw[c] + gb[c];
  xnt[idx] = (h16)v;
}

__global__ void cvt_f32_f16(const float* __restrict__ in, h16* __restrict__ out, int n) {
  int idx = blockIdx.x * blockDim.x + threadIdx.x;
  if (idx < n) out[idx] = (h16)in[idx];
}

// biasd[h][d+2047] = rel_bias[bucket(d)][h] * sqrt(dh)
__global__ void bias_precompute(const float* __restrict__ rel_bias,
                                float* __restrict__ biasd) {
  int idx = blockIdx.x * blockDim.x + threadIdx.x;
  if (idx >= NHEAD * 4095) return;
  int h = idx / 4095;
  int di = idx - h * 4095;
  int d = di - 2047;          // d = s - t
  int nn = -d;                // n = t - s
  int ret = (nn < 0) ? 16 : 0;
  int na = nn < 0 ? -nn : nn;
  int val;
  if (na < 8) {
    val = na;
  } else {
    int vl = 8 + (int)(logf((float)na * 0.125f) * (8.0f / logf(8.0f)));
    val = vl < 15 ? vl : 15;
  }
  biasd[h * BIAS_PITCH + di] = rel_bias[(ret + val) * NHEAD + h] * 8.0f;
}

// ---------------------------------------------------------------------------
// QKV GEMM: qkv[o][l] = sum_c wq[o][c]*xnt[l][c] + b_qkv[o].
// Block = 4 waves; wave computes 16(M) x 64(N): one A frag feeds 4 WMMAs/K-step.
// K-loop unrolled x2 so next step's b128 loads overlap current WMMAs.
// ---------------------------------------------------------------------------
__global__ void qkv_gemm(const h16* __restrict__ wq, const h16* __restrict__ xnt,
                         const float* __restrict__ b_qkv, h16* __restrict__ qkv_ol,
                         h16* __restrict__ qkv_t) {
  int wave = threadIdx.x >> 5;
  int lane = threadIdx.x & 31;
  int n0 = blockIdx.x * 64;
  int m0 = blockIdx.y * 64 + wave * 16;
  v8f acc[4] = {};
  #pragma unroll 2
  for (int k0 = 0; k0 < C_DIM; k0 += 32) {
    v16h a = load_A_frag(wq + (size_t)m0 * C_DIM + k0, C_DIM);
    #pragma unroll
    for (int j = 0; j < 4; ++j) {
      v16h b = load_B_frag(xnt + (size_t)(n0 + j * 16) * C_DIM + k0, C_DIM);
      acc[j] = wmma_f16f32(a, b, acc[j]);
    }
  }
  int n = lane & 15, hi = lane >> 4;
  #pragma unroll
  for (int j = 0; j < 4; ++j) {
    int l = n0 + j * 16 + n;
    v8h pack;
    #pragma unroll
    for (int r = 0; r < 8; ++r) {
      int o = m0 + hi * 8 + r;
      h16 hv = (h16)(acc[j][r] + b_qkv[o]);
      qkv_ol[(size_t)o * L_DIM + l] = hv;
      pack[r] = hv;
    }
    *(v8h*)(qkv_t + (size_t)l * OC3 + m0 + hi * 8) = pack;
  }
}

// ---------------------------------------------------------------------------
// Flash attention per head. Block = 4 waves, each owns 16 t-rows.
// K/V s-chunks (32 wide) staged into LDS by TDM (double buffered); thread 0's
// wave issues, TENSORcnt + barriers synchronize. P goes through wave-private
// LDS to convert C-layout f32 -> A-layout f16.
// ---------------------------------------------------------------------------
__global__ void attn_flash(const h16* __restrict__ qkvt, const h16* __restrict__ qkvol,
                           const float* __restrict__ biasd, h16* __restrict__ aot) {
  __shared__ __align__(16) h16 ldsK[2][32 * 64];   // [s][c], row stride 64
  __shared__ __align__(16) h16 ldsV[2][64 * 32];   // [c][s], row stride 32
  __shared__ __align__(16) h16 ldsP[4 * 16 * 32];
  int wave = threadIdx.x >> 5;
  int lane = threadIdx.x & 31;
  int n = lane & 15, hi = lane >> 4;
  int h = blockIdx.y;
  int t0 = blockIdx.x * 64 + wave * 16;

  const h16* kglob = qkvt + h * 192 + 64;                      // row s: +s*3072, 64 halfs
  const h16* vglob = qkvol + (size_t)(h * 192 + 128) * L_DIM;  // row c: +c*2048

  // Q fragment (A operand): A[t][c] = qkvt[t][h*192+c]
  const h16* qbase = qkvt + (size_t)t0 * OC3 + h * 192;
  v16h aq0 = load_A_frag(qbase, OC3);
  v16h aq1 = load_A_frag(qbase + 32, OC3);

  v8f o0 = {}, o1 = {}, o2 = {}, o3 = {};
  float mrow[8], lrow[8];
  #pragma unroll
  for (int r = 0; r < 8; ++r) { mrow[r] = -3.0e38f; lrow[r] = 0.f; }

  const float* bias_h = biasd + h * BIAS_PITCH + 2047 - (t0 + hi * 8);
  h16* pw = ldsP + wave * (16 * 32);

  // Prologue: issue chunks 0 and 1 (2 TDM ops each).
  if (threadIdx.x == 0) {
    tdm_load_2d((unsigned)(uintptr_t)&ldsK[0][0], kglob, 64, 32, OC3);
    tdm_load_2d((unsigned)(uintptr_t)&ldsV[0][0], vglob, 32, 64, L_DIM);
    tdm_load_2d((unsigned)(uintptr_t)&ldsK[1][0], kglob + (size_t)32 * OC3, 64, 32, OC3);
    tdm_load_2d((unsigned)(uintptr_t)&ldsV[1][0], vglob + 32, 32, 64, L_DIM);
  }

  for (int i = 0; i < L_DIM / 32; ++i) {
    int s0 = i * 32;
    // chunk i resident once TENSORcnt <= 2 (in-order completion); chunk i+1 in flight
    if (i < (L_DIM / 32 - 1)) {
      __builtin_amdgcn_s_wait_tensorcnt(2);
    } else {
      __builtin_amdgcn_s_wait_tensorcnt(0);
    }
    __syncthreads();
    const h16* kb = &ldsK[i & 1][0];
    const h16* vb = &ldsV[i & 1][0];

    // ---- scores: two 16x16 tiles from LDS K
    v8f sa = {}, sb = {};
    {
      v16h bk0 = load_B_frag(kb, 64);
      v16h bk1 = load_B_frag(kb + 32, 64);
      sa = wmma_f16f32(aq0, bk0, sa);
      sa = wmma_f16f32(aq1, bk1, sa);
      v16h bk2 = load_B_frag(kb + 16 * 64, 64);
      v16h bk3 = load_B_frag(kb + 16 * 64 + 32, 64);
      sb = wmma_f16f32(aq0, bk2, sb);
      sb = wmma_f16f32(aq1, bk3, sb);
    }
    // ---- scale + bias + online softmax per row r (C layout: row = r+hi*8)
    #pragma unroll
    for (int r = 0; r < 8; ++r) {
      float va = sa[r] * 0.125f + bias_h[s0 + n - r];
      float vb = sb[r] * 0.125f + bias_h[s0 + 16 + n - r];
      float mx = fmaxf(va, vb);
      #pragma unroll
      for (int sft = 1; sft < 16; sft <<= 1) mx = fmaxf(mx, __shfl_xor(mx, sft, 32));
      float mnew = fmaxf(mrow[r], mx);
      float al = __expf(mrow[r] - mnew);
      float ea = __expf(va - mnew);
      float eb = __expf(vb - mnew);
      float rs = ea + eb;
      #pragma unroll
      for (int sft = 1; sft < 16; sft <<= 1) rs += __shfl_xor(rs, sft, 32);
      lrow[r] = lrow[r] * al + rs;
      mrow[r] = mnew;
      o0[r] *= al; o1[r] *= al; o2[r] *= al; o3[r] *= al;
      pw[(r + hi * 8) * 32 + n]      = (h16)ea;
      pw[(r + hi * 8) * 32 + 16 + n] = (h16)eb;
    }
    asm volatile("s_wait_dscnt 0" ::: "memory");
    // ---- O += P(16x32) x V(32x64), V from LDS
    v16h ap = load_A_frag(pw, 32);
    o0 = wmma_f16f32(ap, load_B_frag(vb, 32), o0);
    o1 = wmma_f16f32(ap, load_B_frag(vb + 16 * 32, 32), o1);
    o2 = wmma_f16f32(ap, load_B_frag(vb + 32 * 32, 32), o2);
    o3 = wmma_f16f32(ap, load_B_frag(vb + 48 * 32, 32), o3);

    __syncthreads();  // everyone done reading buf (i&1) before TDM overwrites it
    if (threadIdx.x == 0 && i + 2 < L_DIM / 32) {
      int sn = (i + 2) * 32;
      tdm_load_2d((unsigned)(uintptr_t)&ldsK[i & 1][0], kglob + (size_t)sn * OC3, 64, 32, OC3);
      tdm_load_2d((unsigned)(uintptr_t)&ldsV[i & 1][0], vglob + sn, 32, 64, L_DIM);
    }
  }
  // ---- normalize + write aot[t][h*64 + c] (f16)
  #pragma unroll
  for (int r = 0; r < 8; ++r) {
    int t = t0 + hi * 8 + r;
    float inv = 1.0f / lrow[r];
    h16* op = aot + (size_t)t * C_DIM + h * DH + n;
    op[0]  = (h16)(o0[r] * inv);
    op[16] = (h16)(o1[r] * inv);
    op[32] = (h16)(o2[r] * inv);
    op[48] = (h16)(o3[r] * inv);
  }
}

// ---------------------------------------------------------------------------
// Output projection + residual.
// ---------------------------------------------------------------------------
__global__ void proj_gemm(const h16* __restrict__ wp, const h16* __restrict__ aot,
                          const float* __restrict__ b_proj, const float* __restrict__ x,
                          float* __restrict__ out) {
  int wave = threadIdx.x >> 5;
  int lane = threadIdx.x & 31;
  int n0 = blockIdx.x * 64;
  int m0 = blockIdx.y * 64 + wave * 16;
  v8f acc[4] = {};
  #pragma unroll 2
  for (int k0 = 0; k0 < C_DIM; k0 += 32) {
    v16h a = load_A_frag(wp + (size_t)m0 * C_DIM + k0, C_DIM);
    #pragma unroll
    for (int j = 0; j < 4; ++j) {
      v16h b = load_B_frag(aot + (size_t)(n0 + j * 16) * C_DIM + k0, C_DIM);
      acc[j] = wmma_f16f32(a, b, acc[j]);
    }
  }
  int n = lane & 15, hi = lane >> 4;
  #pragma unroll
  for (int j = 0; j < 4; ++j) {
    int t = n0 + j * 16 + n;
    #pragma unroll
    for (int r = 0; r < 8; ++r) {
      int o = m0 + hi * 8 + r;
      out[(size_t)o * L_DIM + t] = x[(size_t)o * L_DIM + t] + b_proj[o] + acc[j][r];
    }
  }
}

// ---------------------------------------------------------------------------
// Launch
// ---------------------------------------------------------------------------
extern "C" void kernel_launch(void* const* d_in, const int* in_sizes, int n_in,
                              void* d_out, int out_size, void* d_ws, size_t ws_size,
                              hipStream_t stream) {
  const float* x        = (const float*)d_in[0];
  const float* gn_w     = (const float*)d_in[1];
  const float* gn_b     = (const float*)d_in[2];
  const float* w_qkv    = (const float*)d_in[3];
  const float* b_qkv    = (const float*)d_in[4];
  const float* w_proj   = (const float*)d_in[5];
  const float* b_proj   = (const float*)d_in[6];
  const float* rel_bias = (const float*)d_in[7];
  float* out = (float*)d_out;

  char* ws = (char*)d_ws;
  float* mean  = (float*)(ws + 0);          // 32 f32
  float* rstd  = (float*)(ws + 256);        // 32 f32
  float* biasd = (float*)(ws + 512);        // 16*4096 f32 = 256 KiB
  h16* xnt     = (h16*)(ws + 262656);       // 2048x1024 f16 = 4 MiB
  h16* wqh     = (h16*)(ws + 4456960);      // 3072x1024 f16 = 6 MiB
  h16* wph     = (h16*)(ws + 10748416);     // 1024x1024 f16 = 2 MiB
  h16* qkv_ol  = (h16*)(ws + 12845568);     // 3072x2048 f16 = 12 MiB
  h16* qkv_t   = (h16*)(ws + 25428480);     // 2048x3072 f16 = 12 MiB
  h16* aot     = (h16*)(ws + 38011392);     // 2048x1024 f16 = 4 MiB (end ~40.3 MiB)

  gn_stats<<<32, 256, 0, stream>>>(x, mean, rstd);
  gn_apply<<<(C_DIM * L_DIM) / 256, 256, 0, stream>>>(x, gn_w, gn_b, mean, rstd, xnt);
  cvt_f32_f16<<<(OC3 * C_DIM) / 256, 256, 0, stream>>>(w_qkv, wqh, OC3 * C_DIM);
  cvt_f32_f16<<<(C_DIM * C_DIM) / 256, 256, 0, stream>>>(w_proj, wph, C_DIM * C_DIM);
  bias_precompute<<<(NHEAD * 4095 + 255) / 256, 256, 0, stream>>>(rel_bias, biasd);
  qkv_gemm<<<dim3(L_DIM / 64, OC3 / 64), 128, 0, stream>>>(wqh, xnt, b_qkv, qkv_ol, qkv_t);
  attn_flash<<<dim3(L_DIM / 64, NHEAD), 128, 0, stream>>>(qkv_t, qkv_ol, biasd, aot);
  proj_gemm<<<dim3(L_DIM / 64, C_DIM / 64), 128, 0, stream>>>(wph, aot, b_proj, x, out);
}
